// GCN_regressor_46119358824917
// MI455X (gfx1250) — matrix operand order repaired and checked by reading, not compiled, not run on hardware
//
#include <hip/hip_runtime.h>
#include <hip/hip_bf16.h>
#include <math.h>

typedef __bf16 bf16;
typedef __attribute__((ext_vector_type(16))) __bf16 v16bf;
typedef __attribute__((ext_vector_type(8)))  float  v8f;

#define F_DIM 128
#define H_DIM 256

// ----------------- degree / normalization -----------------
__global__ void k_init_deg(float* deg, int n) {
    int i = blockIdx.x * blockDim.x + threadIdx.x;
    if (i < n) deg[i] = 1.0f;             // self-loop contribution
}
__global__ void k_deg_scatter(const int* __restrict__ dst, float* __restrict__ deg, int E) {
    int e = blockIdx.x * blockDim.x + threadIdx.x;
    if (e < E) atomicAdd(&deg[dst[e]], 1.0f);
}
__global__ void k_rsqrt_inplace(float* d, int n) {
    int i = blockIdx.x * blockDim.x + threadIdx.x;
    if (i < n) d[i] = rsqrtf(d[i]);
}

// ----------------- bf16 WMMA GEMM:  Y[M,NOUT] = A[M,K] @ W[K,NOUT] -----------------
// Block: 256 threads = 8 waves. Each wave owns a 16-row stripe and NT=4 adjacent
// 16-col tiles (4 accumulators reuse one A fragment). Block covers 128 rows x 64 cols.
// W column-block is pre-packed in LDS in per-lane WMMA B-fragment order, so the inner
// loop is: 4x global_load_b128 (A) + cvt, then per tile one 32B LDS load + v_wmma.
template <int K, int NOUT>
__global__ __launch_bounds__(256) void k_gemm_bf16(const float* __restrict__ A,
                                                   const float* __restrict__ W,
                                                   float* __restrict__ Y, int M) {
    constexpr int KS = K / 32;           // k-steps of 32
    constexpr int NT = 4;                // 16-col tiles per wave
    constexpr int LOG_KS = (KS == 4) ? 2 : 3;
    __shared__ v16bf sBp[NT * KS * 32];  // packed B fragments, one v16bf per (tile,kstep,lane)

    const int tid   = threadIdx.x;
    const int nbase = blockIdx.y * (NT * 16);

    // pack W[:, nbase .. nbase+63] into per-lane B fragments (bf16)
    // B 32x16 layout: element i of lane -> K = ks*32 + 16*(lane>>4) + i, N = lane&15
    bf16* sraw = (bf16*)sBp;
    for (int idx = tid; idx < NT * KS * 32 * 16; idx += 256) {
        int i    = idx & 15;
        int lane = (idx >> 4) & 31;
        int ks   = (idx >> 9) & (KS - 1);
        int t    = idx >> (9 + LOG_KS);
        int col  = nbase + t * 16 + (lane & 15);
        int krow = ks * 32 + 16 * (lane >> 4) + i;
        sraw[idx] = (bf16)W[krow * NOUT + col];
    }
    __syncthreads();

    const int lane = tid & 31;
    const int wave = tid >> 5;
    const int half = lane >> 4;          // 0: lanes 0-15, 1: lanes 16-31
    const int l16  = lane & 15;
    const int m0   = (blockIdx.x * 8 + wave) * 16;
    if (m0 >= M) return;                 // wave-uniform exit (no further barriers)

    int row = m0 + l16;
    if (row >= M) row = M - 1;           // clamp: D row m depends only on A row m; stores guarded
    const float4* A4 = (const float4*)(A + (size_t)row * K);

    v8f acc[NT] = {};
    for (int ks = 0; ks < KS; ++ks) {
        // A 16x32 layout: lane's elements are K = ks*32 + 16*half + 0..15
        const float4 f0 = A4[ks * 8 + 4 * half + 0];
        const float4 f1 = A4[ks * 8 + 4 * half + 1];
        const float4 f2 = A4[ks * 8 + 4 * half + 2];
        const float4 f3 = A4[ks * 8 + 4 * half + 3];
        v16bf a;
        a[0]  = (bf16)f0.x; a[1]  = (bf16)f0.y; a[2]  = (bf16)f0.z; a[3]  = (bf16)f0.w;
        a[4]  = (bf16)f1.x; a[5]  = (bf16)f1.y; a[6]  = (bf16)f1.z; a[7]  = (bf16)f1.w;
        a[8]  = (bf16)f2.x; a[9]  = (bf16)f2.y; a[10] = (bf16)f2.z; a[11] = (bf16)f2.w;
        a[12] = (bf16)f3.x; a[13] = (bf16)f3.y; a[14] = (bf16)f3.z; a[15] = (bf16)f3.w;
#pragma unroll
        for (int t = 0; t < NT; ++t) {
            v16bf b = sBp[(t * KS + ks) * 32 + lane];
            acc[t] = __builtin_amdgcn_wmma_f32_16x16x32_bf16(false, a, false, b,
                                                             (short)0, acc[t], false, false);
        }
    }
#pragma unroll
    for (int t = 0; t < NT; ++t) {
#pragma unroll
        for (int r = 0; r < 8; ++r) {    // D: M = r + 8*half, N = lane&15
            int m = m0 + r + 8 * half;
            if (m < M) Y[(size_t)m * NOUT + nbase + t * 16 + l16] = acc[t][r];
        }
    }
}

// ----------------- message passing -----------------
// agg[i,:] = Z[i,:] * dinv[i]^2   (self-loop term; also overwrites poisoned ws)
__global__ void k_agg_init(const float* __restrict__ Z, const float* __restrict__ dinv,
                           float* __restrict__ agg, int n, int shift /*log2 D*/) {
    int t = blockIdx.x * blockDim.x + threadIdx.x;
    int total = n << shift;
    if (t < total) {
        float w = dinv[t >> shift];
        agg[t] = Z[t] * w * w;
    }
}
// agg[dst,:] += Z[src,:] * dinv[src]*dinv[dst]  — one thread per (edge, 4 features)
__global__ void k_edge_scatter(const int* __restrict__ src, const int* __restrict__ dst,
                               const float* __restrict__ dinv, const float* __restrict__ Z,
                               float* __restrict__ agg, int E, int shift /*log2(D/4)*/) {
    int t = blockIdx.x * blockDim.x + threadIdx.x;
    if (t >= (E << shift)) return;
    int e  = t >> shift;
    int f4 = (t & ((1 << shift) - 1)) << 2;
    int s = src[e], d = dst[e];
    float norm = dinv[s] * dinv[d];
    int D = 4 << shift;
    const float4 v = *(const float4*)(Z + (size_t)s * D + f4);
    float* o = agg + (size_t)d * D + f4;
    atomicAdd(o + 0, v.x * norm);
    atomicAdd(o + 1, v.y * norm);
    atomicAdd(o + 2, v.z * norm);
    atomicAdd(o + 3, v.w * norm);
}
__global__ void k_bias_act(float* __restrict__ h, const float* __restrict__ b,
                           int total, int mask, int do_tanh) {
    int t = blockIdx.x * blockDim.x + threadIdx.x;
    if (t < total) {
        float v = h[t] + b[t & mask];
        h[t] = do_tanh ? tanhf(v) : v;
    }
}

// ----------------- global feature + pooling -----------------
__global__ void k_xglobal(const float* a, const float* b, const float* c, float* xg, int G) {
    int g = blockIdx.x * blockDim.x + threadIdx.x;
    if (g < G) xg[g] = (a[g] + b[g] + c[g]) * (1.0f / 3.0f);
}
__global__ void k_zero(float* p, int total) {
    int t = blockIdx.x * blockDim.x + threadIdx.x;
    if (t < total) p[t] = 0.0f;
}
// one block (128 threads) per node: accumulate features 0..127, plus global feat + count
__global__ void k_pool(const float* __restrict__ h3, const int* __restrict__ batch,
                       const float* __restrict__ xg, float* __restrict__ pooled,
                       float* __restrict__ counts, int n) {
    int i = blockIdx.x;
    if (i >= n) return;
    int g = batch[i];
    int f = threadIdx.x;
    atomicAdd(&pooled[(size_t)g * 129 + f], h3[(size_t)i * 128 + f]);
    if (f == 0) {
        atomicAdd(&pooled[(size_t)g * 129 + 128], xg[g]);
        atomicAdd(&counts[g], 1.0f);
    }
}
__global__ void k_pool_div(float* __restrict__ pooled, const float* __restrict__ counts, int G) {
    int t = blockIdx.x * blockDim.x + threadIdx.x;
    if (t < G * 129) pooled[t] /= fmaxf(counts[t / 129], 1.0f);
}

// ----------------- tiny MLP head (fp32 scalar) -----------------
__global__ void k_dense(const float* __restrict__ in, const float* __restrict__ W,
                        const float* __restrict__ b, float* __restrict__ out,
                        int M, int K, int Nout, int relu) {
    int t = blockIdx.x * blockDim.x + threadIdx.x;
    if (t >= M * Nout) return;
    int r = t / Nout, c = t - r * Nout;
    float s = b[c];
    for (int k = 0; k < K; ++k) s += in[(size_t)r * K + k] * W[(size_t)k * Nout + c];
    out[t] = relu ? fmaxf(s, 0.0f) : s;
}

extern "C" void kernel_launch(void* const* d_in, const int* in_sizes, int n_in,
                              void* d_out, int out_size, void* d_ws, size_t ws_size,
                              hipStream_t stream) {
    const float* x      = (const float*)d_in[0];
    const int*   ei     = (const int*)d_in[1];
    const int*   batch  = (const int*)d_in[2];
    const float* natoms = (const float*)d_in[3];
    const float* nbonds = (const float*)d_in[4];
    const float* mw     = (const float*)d_in[5];
    const float* W1 = (const float*)d_in[6],  *b1 = (const float*)d_in[7];
    const float* W2 = (const float*)d_in[8],  *b2 = (const float*)d_in[9];
    const float* W3 = (const float*)d_in[10], *b3 = (const float*)d_in[11];
    const float* fW1 = (const float*)d_in[12], *fb1 = (const float*)d_in[13];
    const float* fW2 = (const float*)d_in[14], *fb2 = (const float*)d_in[15];
    const float* fW3 = (const float*)d_in[16], *fb3 = (const float*)d_in[17];
    const float* fW4 = (const float*)d_in[18], *fb4 = (const float*)d_in[19];
    float* out = (float*)d_out;

    const int n = in_sizes[0] / F_DIM;   // 50000
    const int E = in_sizes[1] / 2;       // 800000
    const int G = in_sizes[3];           // 1024
    const int* src = ei;
    const int* dst = ei + E;

    // workspace carve-out (256B-aligned regions)
    float* ws = (float*)d_ws;
    size_t off = 0;
    auto alloc = [&](size_t cnt) { float* p = ws + off; off += (cnt + 63) & ~(size_t)63; return p; };
    float* dinv   = alloc((size_t)n);
    float* buf0   = alloc((size_t)n * H_DIM);
    float* buf1   = alloc((size_t)n * H_DIM);
    float* xg     = alloc((size_t)G);
    float* pooled = alloc((size_t)G * 129);
    float* counts = alloc((size_t)G);
    float* m1     = alloc((size_t)G * 128);
    float* m2     = alloc((size_t)G * 64);
    float* m3     = alloc((size_t)G * 32);
    (void)ws_size; (void)n_in; (void)out_size;

    const int T = 256;
    auto cdiv = [](int a, int b) { return (a + b - 1) / b; };

    // degree + d^-1/2
    k_init_deg     <<<cdiv(n, T), T, 0, stream>>>(dinv, n);
    k_deg_scatter  <<<cdiv(E, T), T, 0, stream>>>(dst, dinv, E);
    k_rsqrt_inplace<<<cdiv(n, T), T, 0, stream>>>(dinv, n);

    // ---- GCN layer 1: 128 -> 256, tanh ----
    k_gemm_bf16<128, 256><<<dim3(cdiv(n, 128), 4), T, 0, stream>>>(x, W1, buf0, n);
    k_agg_init    <<<cdiv(n * 256, T), T, 0, stream>>>(buf0, dinv, buf1, n, 8);
    k_edge_scatter<<<cdiv(E << 6, T), T, 0, stream>>>(src, dst, dinv, buf0, buf1, E, 6);
    k_bias_act    <<<cdiv(n * 256, T), T, 0, stream>>>(buf1, b1, n * 256, 255, 1);

    // ---- GCN layer 2: 256 -> 256, tanh ----
    k_gemm_bf16<256, 256><<<dim3(cdiv(n, 128), 4), T, 0, stream>>>(buf1, W2, buf0, n);
    k_agg_init    <<<cdiv(n * 256, T), T, 0, stream>>>(buf0, dinv, buf1, n, 8);
    k_edge_scatter<<<cdiv(E << 6, T), T, 0, stream>>>(src, dst, dinv, buf0, buf1, E, 6);
    k_bias_act    <<<cdiv(n * 256, T), T, 0, stream>>>(buf1, b2, n * 256, 255, 1);

    // ---- GCN layer 3: 256 -> 128, linear ----
    k_gemm_bf16<256, 128><<<dim3(cdiv(n, 128), 2), T, 0, stream>>>(buf1, W3, buf0, n);
    k_agg_init    <<<cdiv(n * 128, T), T, 0, stream>>>(buf0, dinv, buf1, n, 7);
    k_edge_scatter<<<cdiv(E << 5, T), T, 0, stream>>>(src, dst, dinv, buf0, buf1, E, 5);
    k_bias_act    <<<cdiv(n * 128, T), T, 0, stream>>>(buf1, b3, n * 128, 127, 0);

    // ---- global feature + mean pool to [G,129] ----
    k_xglobal <<<cdiv(G, T), T, 0, stream>>>(natoms, nbonds, mw, xg, G);
    k_zero    <<<cdiv(G * 129, T), T, 0, stream>>>(pooled, G * 129);
    k_zero    <<<cdiv(G, T), T, 0, stream>>>(counts, G);
    k_pool    <<<n, 128, 0, stream>>>(buf1, batch, xg, pooled, counts, n);
    k_pool_div<<<cdiv(G * 129, T), T, 0, stream>>>(pooled, counts, G);

    // ---- head MLP: 129 -> 128 -> 64 -> 32 -> 1 ----
    k_dense<<<cdiv(G * 128, T), T, 0, stream>>>(pooled, fW1, fb1, m1, G, 129, 128, 1);
    k_dense<<<cdiv(G * 64,  T), T, 0, stream>>>(m1,     fW2, fb2, m2, G, 128, 64,  1);
    k_dense<<<cdiv(G * 32,  T), T, 0, stream>>>(m2,     fW3, fb3, m3, G, 64,  32,  1);
    k_dense<<<cdiv(G,       T), T, 0, stream>>>(m3,     fW4, fb4, out, G, 32,  1,  0);
}
